// MambaBlock_43748536877282
// MI455X (gfx1250) — matrix operand order repaired and checked by reading, not compiled, not run on hardware
//
#include <hip/hip_runtime.h>
#include <hip/hip_bf16.h>

// ---------------------------------------------------------------------------
// Problem constants (match reference)
// ---------------------------------------------------------------------------
#define BB   4
#define LL   2048
#define DM   1024
#define DIN  2048
#define NS   16
#define RR   64
#define KK   4
#define HH   1024
#define EPSF 1e-5f
#define ROWS ((size_t)BB * LL)          // 8192 token rows

// ---------------------------------------------------------------------------
// Types for WMMA / TDM
// ---------------------------------------------------------------------------
typedef __attribute__((ext_vector_type(16))) __bf16 v16bf;
typedef __attribute__((ext_vector_type(8)))  float  v8f;
typedef __attribute__((ext_vector_type(4)))  unsigned int v4u;
typedef __attribute__((ext_vector_type(8)))  int v8i;
typedef __attribute__((ext_vector_type(4)))  int v4i;

struct U8x32 { uint4 lo, hi; };   // 32 bytes, bit-castable to v16bf

__device__ __forceinline__ unsigned short f32_to_bf16(float f) {
    unsigned int u = __float_as_uint(f);
    unsigned int r = (u + 0x7FFFu + ((u >> 16) & 1u)) >> 16;   // RNE
    return (unsigned short)r;
}

__device__ __forceinline__ float silu_f(float x) {
    return x / (1.0f + __expf(-x));
}

// ---------------------------------------------------------------------------
// fp32 -> bf16 converter (weights & activations)
// ---------------------------------------------------------------------------
__global__ __launch_bounds__(256) void cvt_bf16_kernel(
    const float* __restrict__ src, unsigned short* __restrict__ dst, size_t n) {
    size_t i = (size_t)blockIdx.x * 256 + threadIdx.x;
    if (i < n) dst[i] = f32_to_bf16(src[i]);
}

// ---------------------------------------------------------------------------
// res = hs + residual ; hn_bf16 = rmsnorm(res) * w      (one block per row)
// ---------------------------------------------------------------------------
__global__ __launch_bounds__(256) void add_norm1_kernel(
    const float* __restrict__ hs, const float* __restrict__ resid,
    const float* __restrict__ w, float* __restrict__ res,
    unsigned short* __restrict__ hnbf) {
    __shared__ float red[256];
    size_t row = blockIdx.x;
    const float* h = hs + row * DM;
    const float* r = resid + row * DM;
    float v[4]; float ss = 0.f;
#pragma unroll
    for (int i = 0; i < 4; ++i) {
        int c = threadIdx.x + i * 256;
        float t = h[c] + r[c];
        v[i] = t; ss += t * t;
    }
    red[threadIdx.x] = ss; __syncthreads();
    for (int s = 128; s > 0; s >>= 1) {
        if ((int)threadIdx.x < s) red[threadIdx.x] += red[threadIdx.x + s];
        __syncthreads();
    }
    float inv = rsqrtf(red[0] * (1.0f / DM) + EPSF);
#pragma unroll
    for (int i = 0; i < 4; ++i) {
        int c = threadIdx.x + i * 256;
        res[row * DM + c] = v[i];
        hnbf[row * DM + c] = f32_to_bf16(v[i] * inv * w[c]);
    }
}

// ---------------------------------------------------------------------------
// res2 = rmsnorm(mix + res) * w2  (fp32 out) ; mixbf = bf16(mix)
// ---------------------------------------------------------------------------
__global__ __launch_bounds__(256) void add_norm2_kernel(
    const float* __restrict__ mix, const float* __restrict__ res,
    const float* __restrict__ w, float* __restrict__ res2,
    unsigned short* __restrict__ mixbf) {
    __shared__ float red[256];
    size_t row = blockIdx.x;
    const float* m = mix + row * DM;
    const float* r = res + row * DM;
    float v[4], mv[4]; float ss = 0.f;
#pragma unroll
    for (int i = 0; i < 4; ++i) {
        int c = threadIdx.x + i * 256;
        float mm = m[c];
        float t = mm + r[c];
        v[i] = t; mv[i] = mm; ss += t * t;
    }
    red[threadIdx.x] = ss; __syncthreads();
    for (int s = 128; s > 0; s >>= 1) {
        if ((int)threadIdx.x < s) red[threadIdx.x] += red[threadIdx.x + s];
        __syncthreads();
    }
    float inv = rsqrtf(red[0] * (1.0f / DM) + EPSF);
#pragma unroll
    for (int i = 0; i < 4; ++i) {
        int c = threadIdx.x + i * 256;
        res2[row * DM + c] = v[i] * inv * w[c];
        mixbf[row * DM + c] = f32_to_bf16(mv[i]);
    }
}

// ---------------------------------------------------------------------------
// Generic bf16 WMMA GEMM:  C[M,N] (f32) = A[M,K] (bf16) * W[N,K]^T (bf16)
// Block tile 128x128, K-step 32, 8 waves (4 row-blocks x 2 col-blocks),
// each wave owns a 32x64 accumulator = 2x4 tiles of v_wmma_f32_16x16x32_bf16.
// In the N-aligned variant, the B (weight) tile is DMA'd into LDS by the
// Tensor Data Mover (one descriptor per K-step, issued by wave 0), using the
// D# pad feature to produce the padded LDS row stride (32 shorts + 8 pad).
// ---------------------------------------------------------------------------
#define BM 128
#define BN 128
#define BK 32
#define LDSS 40   // padded stride in bf16 elements (80B, 16B multiple)

template <bool GUARD_N>
__global__ __launch_bounds__(256) void gemm_bf16_kernel(
    const unsigned short* __restrict__ A,   // M x K, row major
    const unsigned short* __restrict__ W,   // N x K, row major
    float* __restrict__ C,                  // M x N, row major
    int M, int N, int Kd) {
    __shared__ __align__(16) unsigned short sA[BM * LDSS];
    __shared__ __align__(16) unsigned short sB[BN * LDSS];

    const int tid  = threadIdx.x;
    const int wave = tid >> 5;
    const int lane = tid & 31;
    const int wm   = wave & 3;        // row block (x32) within tile
    const int wn   = wave >> 2;       // col block (x64) within tile
    const int rowBase = blockIdx.y * BM;
    const int colBase = blockIdx.x * BN;
    const int half = lane >> 4;       // 0: lanes 0-15, 1: lanes 16-31
    const int l15  = lane & 15;

    v8f acc[2][4] = {};

    const int ldr = tid >> 2;         // 0..63
    const int ldc = (tid & 3) * 8;    // 0,8,16,24

    const unsigned sB_lds = (unsigned)(unsigned long long)(size_t)&sB[0];

    for (int k0 = 0; k0 < Kd; k0 += BK) {
        if (!GUARD_N) {
            // ---- TDM: DMA B tile (BN x BK bf16) into LDS with row padding ----
            if (wave == 0) {
                unsigned long long ga =
                    (unsigned long long)(const void*)(W + (size_t)colBase * Kd + k0);
                v4u g0;
                g0[0] = 1u;                                  // count=1, user D#
                g0[1] = sB_lds;                              // lds_addr
                g0[2] = (unsigned)(ga & 0xFFFFFFFFu);        // global_addr[31:0]
                g0[3] = (unsigned)((ga >> 32) & 0x01FFFFFFu) // global_addr[56:32]
                        | (2u << 30);                        // type = 2 (image)
                unsigned dim0 = (unsigned)(Kd - k0);         // tensor cols remaining
                unsigned dim1 = (unsigned)(N - colBase);     // tensor rows remaining
                v8i g1;
                g1[0] = (int)((1u << 16)      // data_size = 2 bytes
                            | (1u << 20)      // pad_enable
                            | (3u << 22)      // pad_interval: 16 DWORDs (= 32 shorts)
                            | (3u << 25));    // pad_amount:   4 DWORDs (=  8 shorts)
                g1[1] = (int)((dim0 & 0xFFFFu) << 16);                 // dim0 lo16
                g1[2] = (int)((dim0 >> 16) | ((dim1 & 0xFFFFu) << 16));// dim0 hi / dim1 lo
                g1[3] = (int)((dim1 >> 16) | ((unsigned)BK << 16));    // dim1 hi / tile_dim0
                g1[4] = (int)BN;                                       // tile_dim1, tile_dim2=0
                g1[5] = (int)Kd;                                       // dim0_stride lo32
                g1[6] = 0;                                             // stride hi, dim1_stride=0
                g1[7] = 0;
                v4i gz4 = {0, 0, 0, 0};
                v8i gz8 = {0, 0, 0, 0, 0, 0, 0, 0};
                __builtin_amdgcn_tensor_load_to_lds(g0, g1, gz4, gz4, gz8, 0);
            }
        } else {
            // ---- predicated vector-load path (unaligned N, x_proj GEMM) ----
#pragma unroll
            for (int it = 0; it < 2; ++it) {
                int row = ldr + it * 64;
                int n = colBase + row;
                uint4 v = make_uint4(0u, 0u, 0u, 0u);
                if (n < N) v = *(const uint4*)(W + (size_t)n * Kd + k0 + ldc);
                *(uint4*)&sB[row * LDSS + ldc] = v;
            }
        }
        // ---- stage A tile: 128 x 32 (two rows of 64 per thread pass) ----
#pragma unroll
        for (int it = 0; it < 2; ++it) {
            int row = ldr + it * 64;
            const unsigned short* src = A + (size_t)(rowBase + row) * Kd + k0 + ldc;
            *(uint4*)&sA[row * LDSS + ldc] = *(const uint4*)src;
        }
        // prefetch next A K-slab into near caches
        if (k0 + BK < Kd) {
            __builtin_prefetch(A + (size_t)(rowBase + ldr) * Kd + k0 + BK + ldc, 0, 3);
        }
        if (!GUARD_N) {
            if (wave == 0) __builtin_amdgcn_s_wait_tensorcnt(0);
        }
        __syncthreads();

        // ---- fragments per ISA 16-bit layouts ----
        // A 16x32: lanes 0-15 -> row M=l15, K{0..7,16..23}; lanes 16-31 -> K{8..15,24..31}
        v16bf afrag[2], bfrag[4];
#pragma unroll
        for (int m = 0; m < 2; ++m) {
            int row = wm * 32 + m * 16 + l15;
            const unsigned short* p = &sA[row * LDSS + half * 8];
            U8x32 t;
            t.lo = *(const uint4*)p;          // K base .. base+7
            t.hi = *(const uint4*)(p + 16);   // K base+16 .. base+23
            afrag[m] = __builtin_bit_cast(v16bf, t);
        }
        // B 32x16: lanes 0-15 -> col N=l15, K{0..7,16..23}; lanes 16-31 -> K{8..15,24..31}
#pragma unroll
        for (int n = 0; n < 4; ++n) {
            int col = wn * 64 + n * 16 + l15;
            const unsigned short* p = &sB[col * LDSS + half * 8];
            U8x32 t;
            t.lo = *(const uint4*)p;          // K base .. base+7
            t.hi = *(const uint4*)(p + 16);   // K base+16 .. base+23
            bfrag[n] = __builtin_bit_cast(v16bf, t);
        }
#pragma unroll
        for (int m = 0; m < 2; ++m)
#pragma unroll
            for (int n = 0; n < 4; ++n)
                acc[m][n] = __builtin_amdgcn_wmma_f32_16x16x32_bf16(
                    false, afrag[m], false, bfrag[n],
                    (short)0, acc[m][n], false, false);
        __syncthreads();
    }

    // ---- epilogue: C/D layout -> lanes 0-15: M=r, lanes 16-31: M=r+8 ----
#pragma unroll
    for (int m = 0; m < 2; ++m) {
#pragma unroll
        for (int n = 0; n < 4; ++n) {
            int col = colBase + wn * 64 + n * 16 + l15;
            if (!GUARD_N || col < N) {
#pragma unroll
                for (int r = 0; r < 8; ++r) {
                    int row = rowBase + wm * 32 + m * 16 + r + half * 8;
                    C[(size_t)row * N + col] = acc[m][n][r];
                }
            }
        }
    }
}

// ---------------------------------------------------------------------------
// Depthwise causal conv (K=4) + bias + silu over x = xz[:, :DIN]
// out: xconv fp32 and bf16
// ---------------------------------------------------------------------------
__global__ __launch_bounds__(256) void conv_silu_kernel(
    const float* __restrict__ xz, const float* __restrict__ conv_w,
    const float* __restrict__ conv_b, float* __restrict__ xc,
    unsigned short* __restrict__ xcbf) {
    size_t idx = (size_t)blockIdx.x * 256 + threadIdx.x;   // over ROWS*DIN
    if (idx >= ROWS * DIN) return;
    int d = (int)(idx % DIN);
    size_t bl = idx / DIN;
    int l = (int)(bl % LL);
    float s = conv_b[d];
#pragma unroll
    for (int j = 0; j < KK; ++j) {
        int ll = l - (KK - 1) + j;
        if (ll >= 0)
            s += xz[(bl - (KK - 1) + j) * (2 * DIN) + d] * conv_w[d * KK + j];
    }
    float v = silu_f(s);
    xc[idx] = v;
    xcbf[idx] = f32_to_bf16(v);
}

// ---------------------------------------------------------------------------
// Extract dt_low (first RR cols of dbl) as bf16
// ---------------------------------------------------------------------------
__global__ __launch_bounds__(256) void dtlow_kernel(
    const float* __restrict__ dbl, unsigned short* __restrict__ out) {
    size_t idx = (size_t)blockIdx.x * 256 + threadIdx.x;   // over ROWS*RR
    size_t m = idx / RR;
    int c = (int)(idx % RR);
    out[idx] = f32_to_bf16(dbl[m * (RR + 2 * NS) + c]);
}

// ---------------------------------------------------------------------------
// dt = softplus(dt_pre + bias[d])   (in place)
// ---------------------------------------------------------------------------
__global__ __launch_bounds__(256) void softplus_kernel(
    float* __restrict__ dt, const float* __restrict__ bias) {
    size_t idx = (size_t)blockIdx.x * 256 + threadIdx.x;   // over ROWS*DIN
    int d = (int)(idx % DIN);
    float x = dt[idx] + bias[d];
    dt[idx] = (x > 20.f) ? x : log1pf(__expf(x));
}

// ---------------------------------------------------------------------------
// Selective scan. One thread per (b,d) channel; 16-wide state in registers.
// Fused with D-skip and silu(z) gate; emits bf16 y.
// ---------------------------------------------------------------------------
__global__ __launch_bounds__(256) void scan_kernel(
    const float* __restrict__ dt,     // (ROWS, DIN)
    const float* __restrict__ dbl,    // (ROWS, 96): cols 64..79 = B, 80..95 = C
    const float* __restrict__ xconv,  // (ROWS, DIN)
    const float* __restrict__ xz,     // (ROWS, 2*DIN): z at col DIN+d
    const float* __restrict__ A_log,  // (DIN, NS)
    const float* __restrict__ Dp,     // (DIN)
    unsigned short* __restrict__ ybf) // (ROWS, DIN)
{
    __shared__ float sBC[2 * NS];
    int b = blockIdx.x / (DIN / 256);
    int dblk = blockIdx.x % (DIN / 256);
    int d = dblk * 256 + threadIdx.x;

    float a[NS], h[NS];
#pragma unroll
    for (int n = 0; n < NS; ++n) {
        a[n] = -__expf(A_log[(size_t)d * NS + n]);
        h[n] = 0.f;
    }
    float Dv = Dp[d];

    for (int l = 0; l < LL; ++l) {
        size_t row = (size_t)b * LL + l;
        if (threadIdx.x < 2 * NS)
            sBC[threadIdx.x] = dbl[row * (RR + 2 * NS) + RR + threadIdx.x];
        __syncthreads();

        float dtv = dt[row * DIN + d];
        float xv  = xconv[row * DIN + d];
        float zv  = xz[row * (2 * DIN) + DIN + d];
        float dx  = dtv * xv;
        float y = 0.f;
#pragma unroll
        for (int n = 0; n < NS; ++n) {
            h[n] = __expf(dtv * a[n]) * h[n] + dx * sBC[n];
            y += h[n] * sBC[NS + n];
        }
        y += xv * Dv;
        ybf[row * DIN + d] = f32_to_bf16(y * silu_f(zv));
        __syncthreads();
    }
}

// ---------------------------------------------------------------------------
// GLU: yg = fc[:, :H] * silu(fc[:, H:])  -> bf16
// ---------------------------------------------------------------------------
__global__ __launch_bounds__(256) void glu_kernel(
    const float* __restrict__ fc, unsigned short* __restrict__ yg) {
    size_t idx = (size_t)blockIdx.x * 256 + threadIdx.x;   // over ROWS*HH
    size_t m = idx / HH;
    int hc = (int)(idx % HH);
    float y = fc[m * (2 * HH) + hc];
    float g = fc[m * (2 * HH) + HH + hc];
    yg[idx] = f32_to_bf16(y * silu_f(g));
}

// ---------------------------------------------------------------------------
// Host orchestration
// ---------------------------------------------------------------------------
extern "C" void kernel_launch(void* const* d_in, const int* in_sizes, int n_in,
                              void* d_out, int out_size, void* d_ws, size_t ws_size,
                              hipStream_t stream) {
    const float* hs        = (const float*)d_in[0];
    const float* resid     = (const float*)d_in[1];
    const float* norm_w    = (const float*)d_in[2];
    const float* in_proj_w = (const float*)d_in[3];
    const float* conv_w    = (const float*)d_in[4];
    const float* conv_b    = (const float*)d_in[5];
    const float* x_proj_w  = (const float*)d_in[6];
    const float* dt_proj_w = (const float*)d_in[7];
    const float* dt_proj_b = (const float*)d_in[8];
    const float* A_log     = (const float*)d_in[9];
    const float* Dp        = (const float*)d_in[10];
    const float* out_pw    = (const float*)d_in[11];
    const float* norm2_w   = (const float*)d_in[12];
    const float* fc1_w     = (const float*)d_in[13];
    const float* fc2_w     = (const float*)d_in[14];

    float* out  = (float*)d_out;                 // (B,L,DM)
    float* res2 = out + ROWS * DM;               // (B,L,DM)

    // ---- workspace layout (lifetime-aliased) ----
    char* ws = (char*)d_ws;
    size_t off = 0;
    auto alloc = [&](size_t bytes) -> char* {
        char* p = ws + off;
        off += (bytes + 255) & ~(size_t)255;
        return p;
    };
    float*          res_f   = (float*)alloc(ROWS * DM * 4);          // res (long lived)
    unsigned short* act_bf  = (unsigned short*)alloc(ROWS * DIN * 2);// hn -> y -> mix_bf16
    unsigned short* w_in    = (unsigned short*)alloc((size_t)2*DIN*DM*2);
    unsigned short* w_xp    = (unsigned short*)alloc((size_t)(RR+2*NS)*DIN*2);
    unsigned short* w_dt    = (unsigned short*)alloc((size_t)DIN*RR*2);
    unsigned short* w_op    = (unsigned short*)alloc((size_t)DM*DIN*2);
    unsigned short* w_fc1   = (unsigned short*)alloc((size_t)2*HH*DM*2);
    unsigned short* w_fc2   = (unsigned short*)alloc((size_t)DM*HH*2);
    float*          xz_f    = (float*)alloc(ROWS * 2 * DIN * 4);     // xz -> fc (reuse)
    float*          xconv_f = (float*)alloc(ROWS * DIN * 4);
    unsigned short* xcbf    = (unsigned short*)alloc(ROWS * DIN * 2);// xconv_bf16 -> yg
    float*          dbl_f   = (float*)alloc(ROWS * (RR + 2*NS) * 4);
    unsigned short* dtlow   = (unsigned short*)alloc(ROWS * RR * 2);
    float*          dt_f    = (float*)alloc(ROWS * DIN * 4);
    float*          mix_f   = (float*)alloc(ROWS * DM * 4);
    float*          fc_f    = xz_f;                                  // alias: xz dead after scan
    unsigned short* yg_bf   = xcbf;                                  // alias: xcbf dead after GEMM2

    auto cvt = [&](const float* s, unsigned short* d, size_t n) {
        cvt_bf16_kernel<<<(unsigned)((n + 255) / 256), 256, 0, stream>>>(s, d, n);
    };
    auto gemm = [&](const unsigned short* A, const unsigned short* W, float* C,
                    int M, int N, int K) {
        dim3 grid((N + BN - 1) / BN, M / BM);
        if (N % BN == 0)
            gemm_bf16_kernel<false><<<grid, 256, 0, stream>>>(A, W, C, M, N, K);
        else
            gemm_bf16_kernel<true><<<grid, 256, 0, stream>>>(A, W, C, M, N, K);
    };

    // 0) convert weights to bf16
    cvt(in_proj_w, w_in,  (size_t)2 * DIN * DM);
    cvt(x_proj_w,  w_xp,  (size_t)(RR + 2 * NS) * DIN);
    cvt(dt_proj_w, w_dt,  (size_t)DIN * RR);
    cvt(out_pw,    w_op,  (size_t)DM * DIN);
    cvt(fc1_w,     w_fc1, (size_t)2 * HH * DM);
    cvt(fc2_w,     w_fc2, (size_t)DM * HH);

    // 1) res = hs + residual; hn = rmsnorm(res)
    add_norm1_kernel<<<(unsigned)ROWS, 256, 0, stream>>>(hs, resid, norm_w, res_f, act_bf);

    // 2) xz = hn @ in_proj^T
    gemm(act_bf, w_in, xz_f, (int)ROWS, 2 * DIN, DM);

    // 3) causal depthwise conv + silu
    conv_silu_kernel<<<(unsigned)(ROWS * DIN / 256), 256, 0, stream>>>(
        xz_f, conv_w, conv_b, xconv_f, xcbf);

    // 4) dbl = x @ x_proj^T  (N = 96, guarded variant)
    gemm(xcbf, w_xp, dbl_f, (int)ROWS, RR + 2 * NS, DIN);

    // 5) dt_low bf16, dt = softplus(dt_low @ dt_proj^T + b)
    dtlow_kernel<<<(unsigned)(ROWS * RR / 256), 256, 0, stream>>>(dbl_f, dtlow);
    gemm(dtlow, w_dt, dt_f, (int)ROWS, DIN, RR);
    softplus_kernel<<<(unsigned)(ROWS * DIN / 256), 256, 0, stream>>>(dt_f, dt_proj_b);

    // 6) selective scan -> y (bf16, fused D-skip + silu(z) gate)
    scan_kernel<<<BB * (DIN / 256), 256, 0, stream>>>(
        dt_f, dbl_f, xconv_f, xz_f, A_log, Dp, act_bf);

    // 7) mix = y @ out_proj^T
    gemm(act_bf, w_op, mix_f, (int)ROWS, DM, DIN);

    // 8) res2 = rmsnorm(mix + res); mix_bf16 for MLP
    add_norm2_kernel<<<(unsigned)ROWS, 256, 0, stream>>>(mix_f, res_f, norm2_w, res2, act_bf);

    // 9) fc = mix @ fc1^T ; GLU ; out = yg @ fc2^T
    gemm(act_bf, w_fc1, fc_f, (int)ROWS, 2 * HH, DM);
    glu_kernel<<<(unsigned)(ROWS * HH / 256), 256, 0, stream>>>(fc_f, yg_bf);
    gemm(yg_bf, w_fc2, out, (int)ROWS, DM, HH);
}
